// BitLinear_9526237462629
// MI455X (gfx1250) — matrix operand order repaired and checked by reading, not compiled, not run on hardware
//
#include <hip/hip_runtime.h>
#include <stdint.h>

// ---------------------------------------------------------------------------
// BitLinear forward for MI455X (gfx1250, wave32, WMMA + TDM)
//   out[b,s,o] = scale[o] * sum_k x[b,s,k] * wq[o,k],  wq in {-1,0,1}
// M = B*S = 8192, N = 8192, K = 2048
// ---------------------------------------------------------------------------

typedef __attribute__((ext_vector_type(16))) __bf16 v16bf;
typedef __attribute__((ext_vector_type(8)))  __bf16 v8bf;
typedef __attribute__((ext_vector_type(4)))  __bf16 v4bf;
typedef __attribute__((ext_vector_type(8)))  float  v8f;
typedef __attribute__((ext_vector_type(4)))  unsigned int u32x4;
typedef __attribute__((ext_vector_type(8)))  int    i32x8;
typedef __attribute__((ext_vector_type(4)))  int    i32x4;

union FragAB {
    v16bf v;     // 8 VGPRs, feeds v_wmma_f32_16x16x32_bf16
    v8bf  h[2];  // two 16-byte LDS loads
};

#define K_DIM   2048
#define N_DIM   8192
#define M_DIM   8192
#define BM      128
#define BN      128
#define KTILE   64   // K-chunk per LDS buffer (two WMMA k-steps of 32)
#define KPAD    72   // 144-byte LDS rows: stride 36 banks -> conflict-free b128

#if __has_builtin(__builtin_amdgcn_tensor_load_to_lds) && \
    __has_builtin(__builtin_amdgcn_s_wait_tensorcnt)
#define USE_TDM 1
#else
#define USE_TDM 0
#endif

// ---------------------------------------------------------------------------
// Kernel 1: per-output-row absmean scale + ternary quantize (store as bf16)
// ---------------------------------------------------------------------------
__global__ __launch_bounds__(256) void bitlinear_quant_w(
    const float* __restrict__ w, __bf16* __restrict__ wq,
    float* __restrict__ scales)
{
    const int o   = blockIdx.x;
    const int tid = threadIdx.x;
    const float* row = w + (size_t)o * K_DIM;

    float s = 0.0f;
    #pragma unroll
    for (int i = 0; i < K_DIM / 256; ++i)
        s += fabsf(row[tid + i * 256]);

    __shared__ float red[256];
    red[tid] = s;
    __syncthreads();
    #pragma unroll
    for (int off = 128; off > 0; off >>= 1) {
        if (tid < off) red[tid] += red[tid + off];
        __syncthreads();
    }
    const float scale = fmaxf(red[0] * (1.0f / (float)K_DIM), 1e-5f);
    const float inv   = 1.0f / scale;

    __bf16* orow = wq + (size_t)o * K_DIM;
    #pragma unroll
    for (int i = 0; i < K_DIM / 256; ++i) {
        const int idx = tid + i * 256;
        float q = rintf(row[idx] * inv);           // round half-to-even, like jnp.round
        q = fminf(fmaxf(q, -1.0f), 1.0f);          // ternary
        orow[idx] = (__bf16)q;                     // {-1,0,1} exact in bf16
    }
    if (tid == 0) scales[o] = scale;
}

// ---------------------------------------------------------------------------
// Kernel 2: x f32 -> bf16 (halves GEMM-side activation read traffic)
// ---------------------------------------------------------------------------
__global__ __launch_bounds__(256) void bitlinear_cvt_x(
    const float* __restrict__ x, __bf16* __restrict__ xb)
{
    const size_t i = ((size_t)blockIdx.x * 256 + threadIdx.x) * 4;
    const float4 v = *reinterpret_cast<const float4*>(x + i);
    v4bf o;
    o[0] = (__bf16)v.x; o[1] = (__bf16)v.y;
    o[2] = (__bf16)v.z; o[3] = (__bf16)v.w;
    *reinterpret_cast<v4bf*>(xb + i) = o;
}

// ---------------------------------------------------------------------------
// TDM: one 2D descriptor moves a 128x64 bf16 tile (rows stride K_DIM) into
// LDS, inserting 16B of padding after every 128B row -> KPAD=72 layout.
// ---------------------------------------------------------------------------
#if USE_TDM
__device__ __forceinline__ void tdm_load_tile(const __bf16* gbase,
                                              unsigned int lds_byte_addr)
{
    const unsigned long long ga = (unsigned long long)(uintptr_t)gbase;

    u32x4 g0;
    g0[0] = 1u;                                   // count=1, user descriptor
    g0[1] = lds_byte_addr;                        // D#.lds_addr
    g0[2] = (unsigned int)ga;                     // global_addr[31:0]
    g0[3] = (unsigned int)(ga >> 32)              // global_addr[56:32]
          | (2u << 30);                           // type=2 ("image")

    i32x8 g1;
    g1[0] = (1 << 16)        // data_size = 2 bytes (bf16)
          | (1 << 20)        // pad_enable
          | (4 << 22)        // pad_interval: 32 DWORDs (128B = one 64-elem row)
          | (3 << 25);       // pad_amount:   4 DWORDs (16B) -> 144B LDS rows
    g1[1] = (int)((unsigned)K_DIM << 16);   // tensor_dim0[15:0] = 2048
    g1[2] = (int)((unsigned)M_DIM << 16);   // tensor_dim0 hi=0 | tensor_dim1 lo = 8192
    g1[3] = (int)((unsigned)KTILE << 16);   // tensor_dim1 hi=0 | tile_dim0 = 64
    g1[4] = BM;                             // tile_dim1 = 128 | tile_dim2 = 0
    g1[5] = K_DIM;                          // tensor_dim0_stride[31:0] = 2048
    g1[6] = 0;                              // stride hi | tensor_dim1_stride lo
    g1[7] = 0;

    const i32x4 zero4 = {0, 0, 0, 0};            // 2D tensor: groups 2/3 unused
    const i32x8 zero8 = {0, 0, 0, 0, 0, 0, 0, 0};
    // amdgpu-toolchain (clang-23) 6-arg form:
    // (u32x4 g0, i32x8 g1, i32x4 g2, i32x4 g3, i32x8 extra, i32 cpol)
    __builtin_amdgcn_tensor_load_to_lds(g0, g1, zero4, zero4, zero8, 0);
}
#endif

// ---------------------------------------------------------------------------
// Kernel 3: GEMM via v_wmma_f32_16x16x32_bf16.
// Block: 128x128 output tile, 256 threads = 8 waves (4 in M x 2 in N).
// Wave: 32x64 tile = 2x4 WMMA accumulators. K double-buffered in LDS,
// filled by the Tensor Data Mover (wave 0 drives it, TENSORcnt-tracked).
// 16 WMMAs per workgroup barrier (two k=32 sub-steps per 64-wide buffer).
// ---------------------------------------------------------------------------
__global__ __launch_bounds__(256) void bitlinear_gemm(
    const __bf16* __restrict__ xb,   // [M, K] bf16 activations
    const __bf16* __restrict__ wq,   // [N, K] bf16 ternary weights
    const float*  __restrict__ scales,
    float* __restrict__ out)         // [M, N] f32
{
    __shared__ __align__(16) __bf16 As[2][BM][KPAD];
    __shared__ __align__(16) __bf16 Bs[2][BN][KPAD];

    const int tid   = threadIdx.x;
    const int lane  = tid & 31;
    const int wave  = tid >> 5;
    const int waveM = wave & 3;   // 0..3
    const int waveN = wave >> 2;  // 0..1
    const int m0 = blockIdx.y * BM;
    const int n0 = blockIdx.x * BN;

    v8f acc[2][4];
    #pragma unroll
    for (int tm = 0; tm < 2; ++tm)
        #pragma unroll
        for (int tn = 0; tn < 4; ++tn)
            acc[tm][tn] = (v8f)(0.0f);

    // fragment addressing per documented CDNA5 WMMA VGPR layouts
    const int fr = lane & 15;         // A row / B column within 16x16 tile
    const int kA = (lane >> 4) << 3;  // A: lanes 16-31 hold K+8 .. (+24)
    const int kB = (lane >> 4) << 3;  // B: lanes 16-31 hold K+8 .. (+24)
    const int NK = K_DIM / KTILE;     // 32 iterations

#if USE_TDM
    const __bf16* gAt = xb + (size_t)m0 * K_DIM;   // A tile base (row m0, k=0)
    const __bf16* gBt = wq + (size_t)n0 * K_DIM;   // B tile base (row n0, k=0)
    const unsigned int ldsA0 = (unsigned int)(uintptr_t)&As[0][0][0];
    const unsigned int ldsA1 = (unsigned int)(uintptr_t)&As[1][0][0];
    const unsigned int ldsB0 = (unsigned int)(uintptr_t)&Bs[0][0][0];
    const unsigned int ldsB1 = (unsigned int)(uintptr_t)&Bs[1][0][0];

    if (wave == 0) {                               // one DMA pair per block
        tdm_load_tile(gAt, ldsA0);
        tdm_load_tile(gBt, ldsB0);
        __builtin_amdgcn_s_wait_tensorcnt(0);
    }
    __syncthreads();
#else
    const int srow  = tid >> 1;
    const int shalf = (tid & 1) * 32;
    const __bf16* gA = xb + (size_t)(m0 + srow) * K_DIM + shalf;
    const __bf16* gB = wq + (size_t)(n0 + srow) * K_DIM + shalf;
    auto stage = [&](int buf, int kc) {
        const uint4* a = reinterpret_cast<const uint4*>(gA + kc);
        uint4* da = reinterpret_cast<uint4*>(&As[buf][srow][shalf]);
        da[0] = a[0]; da[1] = a[1]; da[2] = a[2]; da[3] = a[3];
        const uint4* b = reinterpret_cast<const uint4*>(gB + kc);
        uint4* db = reinterpret_cast<uint4*>(&Bs[buf][srow][shalf]);
        db[0] = b[0]; db[1] = b[1]; db[2] = b[2]; db[3] = b[3];
    };
    stage(0, 0);
    __syncthreads();
#endif

    for (int kt = 0; kt < NK; ++kt) {
        const int cur = kt & 1;

#if USE_TDM
        if (wave == 0 && kt + 1 < NK) {            // prefetch next K-chunk via TDM
            const int kn = (kt + 1) * KTILE;
            tdm_load_tile(gAt + kn, cur ? ldsA0 : ldsA1);
            tdm_load_tile(gBt + kn, cur ? ldsB0 : ldsB1);
        }
#else
        if (kt + 1 < NK) {
            stage(cur ^ 1, (kt + 1) * KTILE);
            if (kt + 2 < NK) {
                __builtin_prefetch(gA + (kt + 2) * KTILE, 0, 3);
                __builtin_prefetch(gB + (kt + 2) * KTILE, 0, 3);
            }
        }
#endif

        #pragma unroll
        for (int s = 0; s < 2; ++s) {              // two k=32 sub-steps per buffer
            const int ks = s * 32;
            FragAB afrag[2], bfrag[4];
            #pragma unroll
            for (int tm = 0; tm < 2; ++tm) {
                const int rl = waveM * 32 + tm * 16 + fr;
                afrag[tm].h[0] = *reinterpret_cast<const v8bf*>(&As[cur][rl][ks + kA]);
                afrag[tm].h[1] = *reinterpret_cast<const v8bf*>(&As[cur][rl][ks + kA + 16]);
            }
            #pragma unroll
            for (int tn = 0; tn < 4; ++tn) {
                const int cl = waveN * 64 + tn * 16 + fr;
                bfrag[tn].h[0] = *reinterpret_cast<const v8bf*>(&Bs[cur][cl][ks + kB]);
                bfrag[tn].h[1] = *reinterpret_cast<const v8bf*>(&Bs[cur][cl][ks + kB + 16]);
            }

            #pragma unroll
            for (int tm = 0; tm < 2; ++tm)
                #pragma unroll
                for (int tn = 0; tn < 4; ++tn)
                    acc[tm][tn] = __builtin_amdgcn_wmma_f32_16x16x32_bf16(
                        /*neg_a=*/false, afrag[tm].v,
                        /*neg_b=*/false, bfrag[tn].v,
                        /*c_mod=*/(short)0, acc[tm][tn],
                        /*reuse_a=*/false, /*reuse_b=*/false);
        }

#if USE_TDM
        if (wave == 0 && kt + 1 < NK)
            __builtin_amdgcn_s_wait_tensorcnt(0);  // next buffer landed
#endif
        __syncthreads();
    }

    // Epilogue: C/D layout -> VGPR r holds M=r (lanes 0-15) or M=8+r (16-31),
    // N = lane%16. Apply per-output-feature scale here.
    #pragma unroll
    for (int tn = 0; tn < 4; ++tn) {
        const int col = n0 + waveN * 64 + tn * 16 + fr;
        const float s = scales[col];
        #pragma unroll
        for (int tm = 0; tm < 2; ++tm) {
            const int mbase = m0 + waveM * 32 + tm * 16 + ((lane >> 4) << 3);
            #pragma unroll
            for (int r = 0; r < 8; ++r)
                out[(size_t)(mbase + r) * N_DIM + col] = acc[tm][tn][r] * s;
        }
    }
}

// ---------------------------------------------------------------------------
// Launch
// ---------------------------------------------------------------------------
extern "C" void kernel_launch(void* const* d_in, const int* in_sizes, int n_in,
                              void* d_out, int out_size, void* d_ws, size_t ws_size,
                              hipStream_t stream)
{
    const float* x = (const float*)d_in[0];   // [4, 2048, 2048] f32
    const float* w = (const float*)d_in[1];   // [8192, 2048]  f32
    float* out = (float*)d_out;               // [4, 2048, 8192] f32

    char* ws = (char*)d_ws;
    __bf16* wq     = (__bf16*)ws;                                   // 32 MB
    __bf16* xb     = (__bf16*)(ws + (size_t)32 * 1024 * 1024);      // 32 MB
    float*  scales = (float*)(ws + (size_t)64 * 1024 * 1024);       // 32 KB

    bitlinear_quant_w<<<N_DIM, 256, 0, stream>>>(w, wq, scales);
    bitlinear_cvt_x<<<(M_DIM * K_DIM) / (256 * 4), 256, 0, stream>>>(x, xb);

    dim3 grid(N_DIM / BN, M_DIM / BM);  // 64 x 64
    bitlinear_gemm<<<grid, 256, 0, stream>>>(xb, wq, scales, out);
}
